// RecurrentLDGNNetwork_19636590478130
// MI455X (gfx1250) — matrix operand order repaired and checked by reading, not compiled, not run on hardware
//
#include <hip/hip_runtime.h>
#include <stddef.h>

// ---------------------------------------------------------------------------
// Problem constants (from reference)
// ---------------------------------------------------------------------------
#define NN   50000
#define EE   800000
#define ETOT (EE + NN)
#define INF_ 64
#define HIDD 128
#define HH   4
#define HC   512
#define LD   1152
#define BB   64
#define TT   16
#define G4   (4 * LD)   // 4608

// ---------------------------------------------------------------------------
// Types for WMMA / staging
// ---------------------------------------------------------------------------
typedef __attribute__((ext_vector_type(16))) __bf16 v16bf;
typedef __attribute__((ext_vector_type(8)))  float  v8f;
typedef __attribute__((ext_vector_type(4)))  unsigned int u32x4;
typedef __attribute__((ext_vector_type(8)))  unsigned int u32x8;

union Frag16 { v16bf v; u32x4 u[2]; };
union Sh16   { u32x4 q[2]; unsigned short s[16]; };

__device__ __forceinline__ unsigned short f32_to_bf16(float f) {
    unsigned u = __float_as_uint(f);
    unsigned r = u + 0x7FFFu + ((u >> 16) & 1u);   // round-to-nearest-even
    return (unsigned short)(r >> 16);
}

__device__ __forceinline__ unsigned fenc(float f) {
    unsigned u = __float_as_uint(f);
    return (u & 0x80000000u) ? ~u : (u | 0x80000000u);
}
__device__ __forceinline__ float fdec(unsigned k) {
    unsigned u = (k & 0x80000000u) ? (k ^ 0x80000000u) : ~k;
    return __uint_as_float(u);
}
__device__ __forceinline__ int finitef(float x) {
    return (__float_as_uint(x) & 0x7F800000u) != 0x7F800000u;
}

// ---------------------------------------------------------------------------
// Elementwise helpers
// ---------------------------------------------------------------------------
__global__ void k_cvt_bf16(const float* __restrict__ a, unsigned short* __restrict__ b, size_t n) {
    size_t i = (size_t)blockIdx.x * blockDim.x + threadIdx.x;
    if (i < n) b[i] = f32_to_bf16(a[i]);
}

__global__ void k_zero_u32(unsigned* __restrict__ p, size_t n) {
    size_t i = (size_t)blockIdx.x * blockDim.x + threadIdx.x;
    if (i < n) p[i] = 0u;
}

// out = relu(agg + bias[i % 512]); write f32 and/or bf16
__global__ void k_bias_relu(const float* __restrict__ agg, const float* __restrict__ bias,
                            float* __restrict__ of, unsigned short* __restrict__ obf, size_t n) {
    size_t i = (size_t)blockIdx.x * blockDim.x + threadIdx.x;
    if (i >= n) return;
    float v = agg[i] + bias[i & (HC - 1)];
    v = fmaxf(v, 0.0f);
    if (of)  of[i]  = v;
    if (obf) obf[i] = f32_to_bf16(v);
}

// ---------------------------------------------------------------------------
// Generic bf16 WMMA GEMM: C[M,N] = A[M,K] * B[K,N] (+bias) (relu?)
// Block: 256 threads = 8 waves. Tile: 64 x 128, BK = 32.
// Requires: N % 128 == 0, K % 32 == 0. M guarded.
// Staging on gfx1250:
//   A-tile: per-lane GLOBAL_LOAD_ASYNC_TO_LDS_B128 (ASYNCcnt)
//   B-tile: TDM tensor_load_to_lds (TENSORcnt) into raw [k][n],
//           then LDS->LDS transpose into padded Bs[n][k] so fragment
//           loads are straight ds_load_b128 feeding v_wmma.
// ---------------------------------------------------------------------------
#define SA 40   // padded LDS row stride (halves): 80B, 16B-aligned rows

__global__ __launch_bounds__(256) void k_gemm(
    const unsigned short* __restrict__ A, const unsigned short* __restrict__ B,
    const float* __restrict__ bias, float* __restrict__ Cf,
    unsigned short* __restrict__ Cbf, int M, int Nn, int K, int relu)
{
    __shared__ __align__(16) unsigned short As[64 * SA];
    __shared__ __align__(16) unsigned short Bs[128 * SA];     // transposed [n][k]
#if defined(__gfx1250__)
    __shared__ __align__(16) unsigned short Braw[32 * 128];   // TDM landing [k][n]
#endif

    const int tid  = threadIdx.x;
    const int wave = tid >> 5;
    const int lane = tid & 31;
    const int kh   = lane >> 4;      // lane half (0/1)
    const int lr   = lane & 15;
    const int mb   = blockIdx.y * 64;
    const int nb   = blockIdx.x * 128;
    const int mt   = wave >> 1;            // m-tile 0..3 (16 rows each)
    const int ng   = (wave & 1) * 64;      // wave's 64-col slab (4 n-tiles)

    v8f acc[4] = {};

    const int ksteps = K >> 5;
    for (int kb = 0; kb < ksteps; ++kb) {
        const int k0 = kb << 5;

#if defined(__gfx1250__)
        // --- TDM: wave 0 issues the B tile (32 rows x 128 cols of 2B) ---
        if (tid < 32) {
            unsigned long long ga = (unsigned long long)(size_t)(B + (size_t)k0 * Nn + nb);
            u32x4 g0;
            g0[0] = 1u;                                     // count=1, user mode
            g0[1] = (unsigned)(size_t)(&Braw[0]);           // lds_addr
            g0[2] = (unsigned)ga;                           // global_addr[31:0]
            g0[3] = ((unsigned)(ga >> 32) & 0x01FFFFFFu)    // global_addr[56:32]
                    | 0x80000000u;                          // type = 2 ("image")
            u32x8 g1;
            g1[0] = 0x00010000u;                            // data_size = 1 (2 bytes)
            g1[1] = ((unsigned)Nn & 0xFFFFu) << 16;         // tensor_dim0 low16
            g1[2] = (((unsigned)Nn >> 16) & 0xFFFFu)        // tensor_dim0 high16
                    | (32u << 16);                          // tensor_dim1 low16 = 32
            g1[3] = (128u << 16);                           // tile_dim0 = 128
            g1[4] = 32u;                                    // tile_dim1 = 32, tile_dim2 = 0
            g1[5] = (unsigned)Nn;                           // tensor_dim0_stride low32
            g1[6] = 0u;
            g1[7] = 0u;
            asm volatile("tensor_load_to_lds %0, %1" :: "s"(g0), "s"(g1) : "memory");
            __builtin_amdgcn_s_wait_tensorcnt(0);
        }
        // --- async A tile: each lane moves 16B global -> LDS directly ---
        {
            int r = tid >> 2;
            int c = (tid & 3) * 8;
            int gr = mb + r;
            unsigned short* lp = As + r * SA + c;
            if (gr < M) {
                unsigned lo = (unsigned)(size_t)lp;
                unsigned long long ga =
                    (unsigned long long)(size_t)(A + (size_t)gr * K + k0 + c);
                asm volatile("global_load_async_to_lds_b128 %0, %1, off"
                             :: "v"(lo), "v"(ga) : "memory");
            } else {
                *(u32x4*)lp = (u32x4)0;   // pad rows beyond M
            }
        }
        asm volatile("s_wait_asynccnt 0" ::: "memory");
        __syncthreads();                       // Braw + As visible to all waves
        // --- LDS->LDS transpose: Braw[k][n] -> Bs[n][k] (padded) ---
        {
            int k = tid >> 3;                  // 0..31
            int c = (tid & 7) * 16;            // 0..112
            Sh16 t;
            t.q[0] = *(const u32x4*)(Braw + k * 128 + c);
            t.q[1] = *(const u32x4*)(Braw + k * 128 + c + 8);
#pragma unroll
            for (int i = 0; i < 16; ++i) Bs[(c + i) * SA + k] = t.s[i];
        }
#else
        // --- fallback staging (host pass / non-gfx1250) ---
        {
            int r = tid >> 2;
            int c = (tid & 3) * 8;
            u32x4 val = (u32x4)0;
            int gr = mb + r;
            if (gr < M) val = *(const u32x4*)(A + (size_t)gr * K + k0 + c);
            *(u32x4*)(As + r * SA + c) = val;
        }
        {
            int k = tid >> 3;
            int c = (tid & 7) * 16;
            const unsigned short* bp = B + (size_t)(k0 + k) * Nn + nb + c;
            Sh16 t;
            t.q[0] = *(const u32x4*)bp;
            t.q[1] = *(const u32x4*)(bp + 8);
#pragma unroll
            for (int i = 0; i < 16; ++i) Bs[(c + i) * SA + k] = t.s[i];
        }
#endif
        if (kb + 1 < ksteps) {
            __builtin_prefetch(A + (size_t)(mb + (tid >> 2)) * K + k0 + 32, 0, 3);
        }
        __syncthreads();

        // --- fragments + WMMA (pure ds_load_b128 feeds) ---
        Frag16 a;
        const unsigned short* ap = As + (mt * 16 + lr) * SA + kh * 8;
        a.u[0] = *(const u32x4*)ap;          // K = kh*8 .. +7
        a.u[1] = *(const u32x4*)(ap + 16);   // K = 16+kh*8 .. +7
#pragma unroll
        for (int nt = 0; nt < 4; ++nt) {
            Frag16 b;
            const unsigned short* bp = Bs + (ng + nt * 16 + lr) * SA + kh * 8;
            b.u[0] = *(const u32x4*)bp;       // K = kh*8 .. +7
            b.u[1] = *(const u32x4*)(bp + 16); // K = 16+kh*8 .. +7
            acc[nt] = __builtin_amdgcn_wmma_f32_16x16x32_bf16(
                false, a.v, false, b.v, (short)0, acc[nt], false, false);
        }
        __syncthreads();
    }

    // --- store C (lane holds N = lane&15; VGPR v holds M = kh*8+v) ---
#pragma unroll
    for (int nt = 0; nt < 4; ++nt) {
        int gn = nb + ng + nt * 16 + lr;
        float bv = bias ? bias[gn] : 0.0f;
#pragma unroll
        for (int v = 0; v < 8; ++v) {
            int gm = mb + mt * 16 + kh * 8 + v;
            if (gm < M) {
                float x = acc[nt][v] + bv;
                if (relu) x = fmaxf(x, 0.0f);
                if (Cf)  Cf[(size_t)gm * Nn + gn]  = x;
                if (Cbf) Cbf[(size_t)gm * Nn + gn] = f32_to_bf16(x);
            }
        }
    }
}

// ---------------------------------------------------------------------------
// GATv2 edge kernels (one wave32 per edge where noted)
// ---------------------------------------------------------------------------
__global__ __launch_bounds__(256) void k_edge_logits(
    const float* __restrict__ xl, const float* __restrict__ xr,
    const float* __restrict__ att, const int* __restrict__ ei,
    float* __restrict__ logits, unsigned* __restrict__ menc)
{
    int e = blockIdx.x * 8 + (threadIdx.x >> 5);
    int lane = threadIdx.x & 31;
    if (e >= ETOT) return;
    int s, d;
    if (e < EE) { s = ei[e]; d = ei[EE + e]; } else { s = e - EE; d = s; }
    const float* ps = xl + (size_t)s * HC;
    const float* pd = xr + (size_t)d * HC;
    float part[4] = {0.f, 0.f, 0.f, 0.f};
#pragma unroll
    for (int k = 0; k < 16; ++k) {
        int idx = lane + k * 32;                       // channel 0..511; head = k>>2
        float v = ps[idx] + pd[idx];
        v = (v > 0.0f) ? v : 0.2f * v;                 // leaky_relu 0.2
        part[k >> 2] += v * att[idx];                  // att flat [h*128+c] == idx
    }
#pragma unroll
    for (int h = 0; h < 4; ++h) {
        float x = part[h];
        for (int off = 16; off > 0; off >>= 1) x += __shfl_xor(x, off, 32);
        if (lane == 0) {
            logits[(size_t)e * 4 + h] = x;
            atomicMax(&menc[d * 4 + h], fenc(x));
        }
    }
}

__global__ void k_edge_exp(float* __restrict__ lg, const unsigned* __restrict__ menc,
                           float* __restrict__ denom, const int* __restrict__ ei)
{
    int e = blockIdx.x * blockDim.x + threadIdx.x;
    if (e >= ETOT) return;
    int d = (e < EE) ? ei[EE + e] : (e - EE);
#pragma unroll
    for (int h = 0; h < 4; ++h) {
        float m = fdec(menc[d * 4 + h]);
        if (!finitef(m)) m = 0.0f;
        float ex = __expf(lg[(size_t)e * 4 + h] - m);
        lg[(size_t)e * 4 + h] = ex;
        atomicAdd(&denom[d * 4 + h], ex);
    }
}

__global__ __launch_bounds__(256) void k_edge_agg(
    const float* __restrict__ ex, const float* __restrict__ denom,
    const float* __restrict__ xl, const int* __restrict__ ei,
    float* __restrict__ agg)
{
    int e = blockIdx.x * 8 + (threadIdx.x >> 5);
    int lane = threadIdx.x & 31;
    if (e >= ETOT) return;
    int s, d;
    if (e < EE) { s = ei[e]; d = ei[EE + e]; } else { s = e - EE; d = s; }
    float al[4];
#pragma unroll
    for (int h = 0; h < 4; ++h)
        al[h] = ex[(size_t)e * 4 + h] / (denom[d * 4 + h] + 1e-16f);
    const float* ps = xl + (size_t)s * HC;
    float* pd = agg + (size_t)d * HC;
#pragma unroll
    for (int k = 0; k < 16; ++k) {
        int idx = lane + k * 32;
        atomicAdd(pd + idx, al[k >> 2] * ps[idx]);
    }
}

// ---------------------------------------------------------------------------
// Sequence gather: feat[i, :] = concat(h0[idx], g1[idx], g2[idx]) -> bf16
// ---------------------------------------------------------------------------
__global__ void k_gather(const float* __restrict__ h0, const float* __restrict__ g1,
                         const float* __restrict__ g2, const int* __restrict__ idx,
                         unsigned short* __restrict__ feat)
{
    int i = blockIdx.x;           // 0..1023
    int node = idx[i];
    for (int j = threadIdx.x; j < LD; j += blockDim.x) {
        float v;
        if (j < HIDD)            v = h0[(size_t)node * HIDD + j];
        else if (j < HIDD + HC)  v = g1[(size_t)node * HC + (j - HIDD)];
        else                     v = g2[(size_t)node * HC + (j - HIDD - HC)];
        feat[(size_t)i * LD + j] = f32_to_bf16(v);
    }
}

// ---------------------------------------------------------------------------
// LSTM cell (gate order i,f,g,o). One thread per (b, j).
// ---------------------------------------------------------------------------
__global__ void k_lstm_cell(const float* __restrict__ Xall, const float* __restrict__ Hmat,
                            const float* __restrict__ bih, const float* __restrict__ bhh,
                            const float* __restrict__ cin, float* __restrict__ cout,
                            float* __restrict__ hf, unsigned short* __restrict__ hbf, int t)
{
    int i = blockIdx.x * blockDim.x + threadIdx.x;
    if (i >= BB * LD) return;
    int b = i / LD, j = i - b * LD;
    const float* xg = Xall + (size_t)(b * TT + t) * G4;
    const float* hg = Hmat + (size_t)b * G4;
    float gi = xg[j]            + hg[j]            + bih[j]            + bhh[j];
    float gf = xg[LD + j]       + hg[LD + j]       + bih[LD + j]       + bhh[LD + j];
    float gg = xg[2 * LD + j]   + hg[2 * LD + j]   + bih[2 * LD + j]   + bhh[2 * LD + j];
    float go = xg[3 * LD + j]   + hg[3 * LD + j]   + bih[3 * LD + j]   + bhh[3 * LD + j];
    float si = 1.0f / (1.0f + __expf(-gi));
    float sf = 1.0f / (1.0f + __expf(-gf));
    float so = 1.0f / (1.0f + __expf(-go));
    float c  = sf * cin[i] + si * tanhf(gg);
    float h  = so * tanhf(c);
    cout[i] = c;
    hf[i]   = h;
    hbf[i]  = f32_to_bf16(h);
}

// ---------------------------------------------------------------------------
// Final dueling head combine: out = q - mean(q) + v  [64,2]
// ---------------------------------------------------------------------------
__global__ void k_final(const float* __restrict__ qh, const float* __restrict__ vh,
                        const float* __restrict__ qW2, const float* __restrict__ qb2,
                        const float* __restrict__ vW2, const float* __restrict__ vb2,
                        float* __restrict__ out)
{
    int b = threadIdx.x;
    if (b >= BB) return;
    float a0 = qb2[0], a1 = qb2[1], av = vb2[0];
    for (int j = 0; j < 128; ++j) {
        float q = qh[b * 128 + j];
        a0 += q * qW2[j * 2];
        a1 += q * qW2[j * 2 + 1];
        av += vh[b * 128 + j] * vW2[j];
    }
    float mean = 0.5f * (a0 + a1);
    out[b * 2]     = a0 - mean + av;
    out[b * 2 + 1] = a1 - mean + av;
}

// ---------------------------------------------------------------------------
// Host orchestration
// ---------------------------------------------------------------------------
static inline size_t cdivz(size_t a, size_t b) { return (a + b - 1) / b; }

extern "C" void kernel_launch(void* const* d_in, const int* in_sizes, int n_in,
                              void* d_out, int out_size, void* d_ws, size_t ws_size,
                              hipStream_t stream) {
    (void)in_sizes; (void)n_in; (void)out_size; (void)ws_size;

    const float* x       = (const float*)d_in[0];
    const int*   ei      = (const int*)d_in[1];
    const int*   indices = (const int*)d_in[2];
    const float* encW1 = (const float*)d_in[5];  const float* encb1 = (const float*)d_in[6];
    const float* encW2 = (const float*)d_in[7];  const float* encb2 = (const float*)d_in[8];
    const float* c1Wl  = (const float*)d_in[9];  const float* c1Wr  = (const float*)d_in[10];
    const float* c1att = (const float*)d_in[11]; const float* c1b   = (const float*)d_in[12];
    const float* c2Wl  = (const float*)d_in[13]; const float* c2Wr  = (const float*)d_in[14];
    const float* c2att = (const float*)d_in[15]; const float* c2b   = (const float*)d_in[16];
    const float* Wih   = (const float*)d_in[17]; const float* Whh   = (const float*)d_in[18];
    const float* bih   = (const float*)d_in[19]; const float* bhh   = (const float*)d_in[20];
    const float* qW1   = (const float*)d_in[21]; const float* qb1   = (const float*)d_in[22];
    const float* qW2   = (const float*)d_in[23]; const float* qb2   = (const float*)d_in[24];
    const float* vW1   = (const float*)d_in[25]; const float* vb1   = (const float*)d_in[26];
    const float* vW2   = (const float*)d_in[27]; const float* vb2   = (const float*)d_in[28];

    // ---- workspace bump allocator (256B aligned) ----
    char* base = (char*)d_ws;
    size_t off = 0;
    auto alloc = [&](size_t bytes) -> void* {
        bytes = (bytes + 255) & ~(size_t)255;
        void* p = base + off; off += bytes; return p;
    };
    unsigned short* xbf    = (unsigned short*)alloc((size_t)NN * INF_ * 2);
    unsigned short* encW1b = (unsigned short*)alloc((size_t)INF_ * 512 * 2);
    unsigned short* encW2b = (unsigned short*)alloc((size_t)512 * HIDD * 2);
    unsigned short* c1Wlb  = (unsigned short*)alloc((size_t)HIDD * HC * 2);
    unsigned short* c1Wrb  = (unsigned short*)alloc((size_t)HIDD * HC * 2);
    unsigned short* c2Wlb  = (unsigned short*)alloc((size_t)HC * HC * 2);
    unsigned short* c2Wrb  = (unsigned short*)alloc((size_t)HC * HC * 2);
    unsigned short* Wihb   = (unsigned short*)alloc((size_t)LD * G4 * 2);
    unsigned short* Whhb   = (unsigned short*)alloc((size_t)LD * G4 * 2);
    unsigned short* qW1b   = (unsigned short*)alloc((size_t)LD * 128 * 2);
    unsigned short* vW1b   = (unsigned short*)alloc((size_t)LD * 128 * 2);
    unsigned short* midbf  = (unsigned short*)alloc((size_t)NN * 512 * 2); // also g1_bf16
    float*          h0f    = (float*)alloc((size_t)NN * HIDD * 4);
    unsigned short* h0bf   = (unsigned short*)alloc((size_t)NN * HIDD * 2);
    float*          xlf    = (float*)alloc((size_t)NN * HC * 4);   // also g2_f32
    float*          xrf    = (float*)alloc((size_t)NN * HC * 4);
    float*          aggf   = (float*)alloc((size_t)NN * HC * 4);
    float*          g1f    = (float*)alloc((size_t)NN * HC * 4);
    float*          elog   = (float*)alloc((size_t)ETOT * 4 * 4);
    unsigned*       menc   = (unsigned*)alloc((size_t)NN * 4 * 4);
    float*          denom  = (float*)alloc((size_t)NN * 4 * 4);
    unsigned short* featbf = (unsigned short*)alloc((size_t)BB * TT * LD * 2);
    float*          Xall   = (float*)alloc((size_t)BB * TT * G4 * 4);
    float*          Hmat   = (float*)alloc((size_t)BB * G4 * 4);
    unsigned short* hbf    = (unsigned short*)alloc((size_t)BB * LD * 2);
    float*          cbuf   = (float*)alloc((size_t)BB * LD * 4);
    float*          qhf    = (float*)alloc((size_t)BB * 128 * 4);
    float*          vhf    = (float*)alloc((size_t)BB * 128 * 4);

    auto cvt = [&](const float* s, unsigned short* d, size_t n) {
        k_cvt_bf16<<<(unsigned)cdivz(n, 256), 256, 0, stream>>>(s, d, n);
    };
    auto zero = [&](void* p, size_t n_u32) {
        k_zero_u32<<<(unsigned)cdivz(n_u32, 256), 256, 0, stream>>>((unsigned*)p, n_u32);
    };
    auto gemm = [&](const unsigned short* A, const unsigned short* B, const float* bias,
                    float* Cf, unsigned short* Cbf, int M, int Nn, int K, int relu) {
        dim3 g((unsigned)(Nn / 128), (unsigned)((M + 63) / 64));
        k_gemm<<<g, 256, 0, stream>>>(A, B, bias, Cf, Cbf, M, Nn, K, relu);
    };

    // ---- bf16 conversions ----
    cvt(x, xbf, (size_t)NN * INF_);
    cvt(encW1, encW1b, (size_t)INF_ * 512);
    cvt(encW2, encW2b, (size_t)512 * HIDD);
    cvt(c1Wl, c1Wlb, (size_t)HIDD * HC);
    cvt(c1Wr, c1Wrb, (size_t)HIDD * HC);
    cvt(c2Wl, c2Wlb, (size_t)HC * HC);
    cvt(c2Wr, c2Wrb, (size_t)HC * HC);
    cvt(Wih, Wihb, (size_t)LD * G4);
    cvt(Whh, Whhb, (size_t)LD * G4);
    cvt(qW1, qW1b, (size_t)LD * 128);
    cvt(vW1, vW1b, (size_t)LD * 128);

    // ---- encoder MLP ----
    gemm(xbf, encW1b, encb1, nullptr, midbf, NN, 512, INF_, 1);
    gemm(midbf, encW2b, encb2, h0f, h0bf, NN, HIDD, 512, 1);

    const unsigned egrid = (unsigned)cdivz(ETOT, 8);

    // ---- GATv2 layer 1 ----
    gemm(h0bf, c1Wlb, nullptr, xlf, nullptr, NN, HC, HIDD, 0);
    gemm(h0bf, c1Wrb, nullptr, xrf, nullptr, NN, HC, HIDD, 0);
    zero(menc, (size_t)NN * 4);
    zero(denom, (size_t)NN * 4);
    zero(aggf, (size_t)NN * HC);
    k_edge_logits<<<egrid, 256, 0, stream>>>(xlf, xrf, c1att, ei, elog, menc);
    k_edge_exp<<<(unsigned)cdivz(ETOT, 256), 256, 0, stream>>>(elog, menc, denom, ei);
    k_edge_agg<<<egrid, 256, 0, stream>>>(elog, denom, xlf, ei, aggf);
    k_bias_relu<<<(unsigned)cdivz((size_t)NN * HC, 256), 256, 0, stream>>>(
        aggf, c1b, g1f, midbf /*g1 bf16*/, (size_t)NN * HC);

    // ---- GATv2 layer 2 (xl/xr buffers reused) ----
    gemm(midbf, c2Wlb, nullptr, xlf, nullptr, NN, HC, HC, 0);
    gemm(midbf, c2Wrb, nullptr, xrf, nullptr, NN, HC, HC, 0);
    zero(menc, (size_t)NN * 4);
    zero(denom, (size_t)NN * 4);
    zero(aggf, (size_t)NN * HC);
    k_edge_logits<<<egrid, 256, 0, stream>>>(xlf, xrf, c2att, ei, elog, menc);
    k_edge_exp<<<(unsigned)cdivz(ETOT, 256), 256, 0, stream>>>(elog, menc, denom, ei);
    k_edge_agg<<<egrid, 256, 0, stream>>>(elog, denom, xlf, ei, aggf);
    // g2 (f32 only) overwrites xlf — xl is dead after aggregation
    k_bias_relu<<<(unsigned)cdivz((size_t)NN * HC, 256), 256, 0, stream>>>(
        aggf, c2b, xlf /*g2_f32*/, nullptr, (size_t)NN * HC);

    // ---- gather sequence features ----
    k_gather<<<BB * TT, 256, 0, stream>>>(h0f, g1f, xlf /*g2*/, indices, featbf);

    // ---- LSTM: hoisted input projection + 16 recurrent steps ----
    gemm(featbf, Wihb, nullptr, Xall, nullptr, BB * TT, G4, LD, 0);
    zero(hbf, (size_t)BB * LD * 2 / 4);
    zero(cbuf, (size_t)BB * LD);

    float* out_f = (float*)d_out;
    float* hlast = out_f + BB * 2;                 // [64,1152]
    float* clast = hlast + (size_t)BB * LD;        // [64,1152]

    for (int t = 0; t < TT; ++t) {
        gemm(hbf, Whhb, nullptr, Hmat, nullptr, BB, G4, LD, 0);
        float* cout_p = (t == TT - 1) ? clast : cbuf;
        k_lstm_cell<<<(unsigned)cdivz((size_t)BB * LD, 256), 256, 0, stream>>>(
            Xall, Hmat, bih, bhh, cbuf, cout_p, hlast, hbf, t);
    }

    // ---- dueling heads ----
    gemm(hbf, qW1b, qb1, qhf, nullptr, BB, 128, LD, 1);
    gemm(hbf, vW1b, vb1, vhf, nullptr, BB, 128, LD, 1);
    k_final<<<1, 64, 0, stream>>>(qhf, vhf, qW2, qb2, vW2, vb2, out_f);
}
